// CoordUpdateWithMsaAndPair_19344532701530
// MI455X (gfx1250) — matrix operand order repaired and checked by reading, not compiled, not run on hardware
//
#include <hip/hip_runtime.h>
#include <hip/hip_bf16.h>
#include <math.h>

typedef _Float16 half_t;
typedef __attribute__((ext_vector_type(16))) _Float16 v16h;
typedef __attribute__((ext_vector_type(8)))  _Float16 v8h;
typedef __attribute__((ext_vector_type(8)))  float    v8f;

static constexpr int LRES = 384;   // residues
static constexpr int NSEQ = 64;    // msa rows
static constexpr int DM   = 384;   // d_msa
static constexpr int DP   = 288;   // d_pair
static constexpr int DN   = 64;    // d_node
static constexpr int DE   = 64;    // d_edge
static constexpr int DSTATE = 32;

__device__ inline float eluf(float x) { return x > 0.f ? x : (expf(x) - 1.f); }

// ---- WMMA helpers (CDNA5 16x16x32 f16, wave32) ----
__device__ inline v8f wmma16(v16h a, v16h b, v8f c) {
#if defined(__gfx1250__)
  return __builtin_amdgcn_wmma_f32_16x16x32_f16(false, a, false, b, (short)0, c, false, false);
#else
  return c;
#endif
}

__device__ inline v16h cat8(v8h lo, v8h hi) {
  return __builtin_shufflevector(lo, hi, 0, 1, 2, 3, 4, 5, 6, 7, 8, 9, 10, 11, 12, 13, 14, 15);
}

// A matrix 16x32 f16 from row-major [16][rs]:
// lane l: M=l&15, half-K-offset 0/8; elements 0-7 = K off..off+7, 8-15 = K 16+off..16+off+7
// => two contiguous 16B loads. Requires rs multiple of 8 halves and 16B-aligned base.
__device__ inline v16h frag_a(const half_t* __restrict__ a, int rs, int kbase, int lane) {
  int M = lane & 15;
  int off = (lane < 16) ? 0 : 8;
  const half_t* p = a + (size_t)M * rs + kbase + off;
  v8h lo = *(const v8h*)p;
  v8h hi = *(const v8h*)(p + 16);
  return cat8(lo, hi);
}

// B matrix 32x16 f16 from N-major [N][rsK] source:
// lane l: N=nbase+(l&15); elements 0-7 = K off..off+7, 8-15 = K 16+off..16+off+7 (off=8*(l>=16)) -> two 16B loads.
__device__ inline v16h frag_bT(const half_t* __restrict__ bT, int rsK, int kbase, int nbase, int lane) {
  int N = nbase + (lane & 15);
  int koff = (lane < 16) ? 0 : 8;
  const half_t* p = bT + (size_t)N * rsK + kbase + koff;
  v8h lo = *(const v8h*)p;
  v8h hi = *(const v8h*)(p + 16);
  return cat8(lo, hi);
}

// ---------------- conversion kernels ----------------
__global__ void cvt_f16_kernel(const float* __restrict__ x, half_t* __restrict__ y, int n) {
  int i = blockIdx.x * blockDim.x + threadIdx.x;
  if (i < n) y[i] = (half_t)x[i];
}
// x row-major [R][C]; y = x^T as [C][R]
__global__ void cvt_f16_T_kernel(const float* __restrict__ x, half_t* __restrict__ y, int R, int C) {
  int i = blockIdx.x * blockDim.x + threadIdx.x;
  if (i < R * C) { int r = i / C, c = i % C; y[(size_t)c * R + r] = (half_t)x[i]; }
}
// x row-major [R][C] (K x N); y N-major padded [Cp][Rp]: y[n*Rp+k] = x[k][n] (zero pad)
__global__ void cvt_f16_padT_kernel(const float* __restrict__ x, half_t* __restrict__ y,
                                    int R, int C, int Rp, int Cp) {
  int i = blockIdx.x * blockDim.x + threadIdx.x;
  if (i < Rp * Cp) {
    int n = i / Rp, k = i % Rp;
    y[i] = (k < R && n < C) ? (half_t)x[(size_t)k * C + n] : (half_t)0.f;
  }
}

// ---------------- row LayerNorm -> f16 ----------------
__global__ void ln_rows_f16_kernel(const float* __restrict__ x, const float* __restrict__ s,
                                   const float* __restrict__ b, half_t* __restrict__ y, int D) {
  __shared__ float red[128];
  int row = blockIdx.x, tid = threadIdx.x;
  const float* xr = x + (size_t)row * D;
  half_t* yr = y + (size_t)row * D;
  float acc = 0.f;
  for (int d = tid; d < D; d += 128) acc += xr[d];
  red[tid] = acc; __syncthreads();
  for (int o = 64; o > 0; o >>= 1) { if (tid < o) red[tid] += red[tid + o]; __syncthreads(); }
  float mean = red[0] / D; __syncthreads();
  acc = 0.f;
  for (int d = tid; d < D; d += 128) { float t = xr[d] - mean; acc += t * t; }
  red[tid] = acc; __syncthreads();
  for (int o = 64; o > 0; o >>= 1) { if (tid < o) red[tid] += red[tid + o]; __syncthreads(); }
  float inv = rsqrtf(red[0] / D + 1e-5f); __syncthreads();
  for (int d = tid; d < D; d += 128)
    yr[d] = (half_t)((xr[d] - mean) * inv * s[d] + b[d]);
}

// ---------------- generic f16 GEMM (64x64 tile / 4 waves) ----------------
// A row-major [M][K] f16; B row-major [K][N] f16 (staged transposed in LDS)
__global__ void gemm_f16_kernel(const half_t* __restrict__ A, const half_t* __restrict__ B,
                                const float* __restrict__ bias, float scale,
                                int M, int N, int K,
                                float* __restrict__ Cf, half_t* __restrict__ Ch) {
  __shared__ half_t As[64 * 32];   // row-major [64][32]
  __shared__ half_t BsT[64 * 32];  // N-major [64 n][32 k]
  int tid = threadIdx.x, lane = tid & 31, wave = tid >> 5;
  int bm = blockIdx.y * 64, bn = blockIdx.x * 64;
  v8f acc[4] = {};
  for (int kk = 0; kk < K; kk += 32) {
    __syncthreads();
    for (int idx = tid; idx < 64 * 32; idx += 128) {
      int r = idx >> 5, c = idx & 31;
      As[idx] = A[(size_t)(bm + r) * K + kk + c];
    }
    for (int idx = tid; idx < 32 * 64; idx += 128) {
      int r = idx >> 6, c = idx & 63; // r = k-local, c = n-local (coalesced global read)
      BsT[c * 32 + r] = B[(size_t)(kk + r) * N + bn + c];
    }
    __syncthreads();
    v16h a = frag_a(As + wave * 16 * 32, 32, 0, lane);
#pragma unroll
    for (int nt = 0; nt < 4; ++nt) {
      v16h bfr = frag_bT(BsT, 32, 0, nt * 16, lane);
      acc[nt] = wmma16(a, bfr, acc[nt]);
    }
  }
  int moff = (lane < 16) ? 0 : 8;
  int nloc = lane & 15;
#pragma unroll
  for (int nt = 0; nt < 4; ++nt) {
    int col = bn + nt * 16 + nloc;
    float bv = bias ? bias[col] : 0.f;
#pragma unroll
    for (int r = 0; r < 8; ++r) {
      int row = bm + wave * 16 + moff + r;
      float v = (acc[nt][r] + bv) * scale;
      if (Cf) Cf[(size_t)row * N + col] = v;
      if (Ch) Ch[(size_t)row * N + col] = (half_t)v;
    }
  }
}

// ---------------- attention pool: logits, softmax over n, pooled msa ----------------
__global__ void att_pool_kernel(const half_t* __restrict__ msa_ln, const float* __restrict__ t,
                                float* __restrict__ pooled) {
  int l = blockIdx.x, n = threadIdx.x; // 64 threads
  __shared__ float logits[64];
  __shared__ float red[64];
  const half_t* mrow = msa_ln + ((size_t)n * LRES + l) * DM;
  const float* tl = t + (size_t)l * DM;
  float acc = 0.f;
  for (int e = 0; e < DM; ++e) acc += (float)mrow[e] * tl[e];
  logits[n] = acc; red[n] = acc; __syncthreads();
  for (int o = 32; o > 0; o >>= 1) { if (n < o) red[n] = fmaxf(red[n], red[n + o]); __syncthreads(); }
  float mx = red[0]; __syncthreads();
  float ex = expf(logits[n] - mx);
  red[n] = ex; __syncthreads();
  for (int o = 32; o > 0; o >>= 1) { if (n < o) red[n] += red[n + o]; __syncthreads(); }
  float inv = 1.f / red[0]; __syncthreads();
  logits[n] = ex * inv; __syncthreads();
  for (int e = n; e < DM; e += 64) {
    float s = 0.f;
    for (int nn = 0; nn < NSEQ; ++nn)
      s += logits[nn] * (float)msa_ln[((size_t)nn * LRES + l) * DM + e];
    pooled[(size_t)l * DM + e] = s;
  }
}

// ---------------- node MLP + LN ----------------
__global__ void node_kernel(const float* __restrict__ pooled, const float* __restrict__ seq,
                            const float* __restrict__ w, const float* __restrict__ bb,
                            const float* __restrict__ ls, const float* __restrict__ lb,
                            float* __restrict__ h0) {
  int l = blockIdx.x, s = threadIdx.x; // 64 threads
  __shared__ float red[64];
  float acc = bb[s];
  const float* pin = pooled + (size_t)l * DM;
  for (int d = 0; d < DM; ++d) acc += pin[d] * w[(size_t)d * DN + s];
  const float* sq = seq + (size_t)l * 21;
  for (int d = 0; d < 21; ++d) acc += sq[d] * w[(size_t)(DM + d) * DN + s];
  acc = eluf(acc);
  red[s] = acc; __syncthreads();
  for (int o = 32; o > 0; o >>= 1) { if (s < o) red[s] += red[s + o]; __syncthreads(); }
  float mean = red[0] / 64.f; __syncthreads();
  float dv = acc - mean;
  red[s] = dv * dv; __syncthreads();
  for (int o = 32; o > 0; o >>= 1) { if (s < o) red[s] += red[s + o]; __syncthreads(); }
  float inv = rsqrtf(red[0] / 64.f + 1e-5f);
  h0[(size_t)l * DN + s] = dv * inv * ls[s] + lb[s];
}

// ---------------- fused pair-LN + edge GEMM(WMMA) + elu + LN -> f16 ----------------
__global__ void edge_kernel(const float* __restrict__ pair, const float* __restrict__ pls,
                            const float* __restrict__ plb, const half_t* __restrict__ ewT,
                            const float* __restrict__ eb, const float* __restrict__ els,
                            const float* __restrict__ elb, half_t* __restrict__ edge) {
  extern __shared__ char smem[];
  half_t* Ap = (half_t*)smem;                         // 64*288 halves
  float*  Out = (float*)(smem + 64 * 288 * 2);        // 64*64 floats
  float*  red = Out + 64 * 64;                        // 128 floats
  int tid = threadIdx.x, lane = tid & 31, wave = tid >> 5;
  size_t row0 = (size_t)blockIdx.x * 64;
  int r = tid >> 1, p = tid & 1;
  const float* prow = pair + (row0 + r) * DP;
  float acc = 0.f;
  for (int e = p * 144; e < p * 144 + 144; ++e) acc += prow[e];
  red[tid] = acc; __syncthreads();
  float mean = (red[2 * r] + red[2 * r + 1]) * (1.f / DP); __syncthreads();
  acc = 0.f;
  for (int e = p * 144; e < p * 144 + 144; ++e) { float d = prow[e] - mean; acc += d * d; }
  red[tid] = acc; __syncthreads();
  float inv = rsqrtf((red[2 * r] + red[2 * r + 1]) / DP + 1e-5f);
  for (int e = p * 144; e < p * 144 + 144; ++e)
    Ap[r * DP + e] = (half_t)((prow[e] - mean) * inv * pls[e] + plb[e]);
  __syncthreads();
  v8f acc4[4] = {};
  for (int kt = 0; kt < DP / 32; ++kt) {
    v16h a = frag_a(Ap + wave * 16 * DP, DP, kt * 32, lane);
#pragma unroll
    for (int nt = 0; nt < 4; ++nt) {
      v16h bfr = frag_bT(ewT, DP, kt * 32, nt * 16, lane);
      acc4[nt] = wmma16(a, bfr, acc4[nt]);
    }
  }
  int moff = (lane < 16) ? 0 : 8;
  int nloc = lane & 15;
#pragma unroll
  for (int nt = 0; nt < 4; ++nt) {
    int col = nt * 16 + nloc;
    float bv = eb[col];
#pragma unroll
    for (int rr = 0; rr < 8; ++rr)
      Out[(wave * 16 + moff + rr) * 64 + col] = eluf(acc4[nt][rr] + bv);
  }
  __syncthreads();
  if (tid < 64) {
    float* o = Out + tid * 64;
    float m = 0.f;
    for (int c = 0; c < 64; ++c) m += o[c];
    m *= (1.f / 64.f);
    float var = 0.f;
    for (int c = 0; c < 64; ++c) { float d = o[c] - m; var += d * d; }
    float iv = rsqrtf(var / 64.f + 1e-5f);
    for (int c = 0; c < 64; ++c)
      edge[(row0 + tid) * 64 + c] = (half_t)((o[c] - m) * iv * els[c] + elb[c]);
  }
}

// ---------------- geometry: r, d_hat, kNN mask via order-statistic bisection ----------------
__global__ void geom_kernel(const float* __restrict__ xyz, const int* __restrict__ aa,
                            float* __restrict__ rmat, float* __restrict__ dhat,
                            float* __restrict__ mask) {
  __shared__ float pd[LRES];
  __shared__ float red[128];
  int i = blockIdx.x, tid = threadIdx.x; // 128 threads
  float cax = xyz[i * 9 + 3], cay = xyz[i * 9 + 4], caz = xyz[i * 9 + 5];
  for (int j = tid; j < LRES; j += 128) {
    float dx = xyz[j * 9 + 3] - cax, dy = xyz[j * 9 + 4] - cay, dz = xyz[j * 9 + 5] - caz;
    float rr = sqrtf(dx * dx + dy * dy + dz * dz + 1e-12f);
    rmat[(size_t)i * LRES + j] = rr;
    float ivr = 1.f / (rr + 1e-8f);
    size_t o3 = ((size_t)i * LRES + j) * 3;
    dhat[o3 + 0] = dx * ivr; dhat[o3 + 1] = dy * ivr; dhat[o3 + 2] = dz * ivr;
    pd[j] = rr + ((i == j) ? 1000.f : 0.f);
  }
  __syncthreads();
  float lo = 0.f, hi = 1e6f;
  for (int it = 0; it < 48; ++it) {
    float mid = 0.5f * (lo + hi);
    float c = 0.f;
    for (int j = tid; j < LRES; j += 128) c += (pd[j] <= mid) ? 1.f : 0.f;
    red[tid] = c; __syncthreads();
    for (int o = 64; o > 0; o >>= 1) { if (tid < o) red[tid] += red[tid + o]; __syncthreads(); }
    float cnt = red[0]; __syncthreads();
    if (cnt >= 64.f) hi = mid; else lo = mid;
  }
  int ai = aa[i];
  for (int j = tid; j < LRES; j += 128) {
    float sep = fabsf((float)(aa[j] - ai)) + ((i == j) ? 999.9f : 0.f);
    mask[(size_t)i * LRES + j] = (pd[j] <= hi || sep < 9.f) ? 1.f : 0.f;
  }
}

__global__ void vinit_kernel(const float* __restrict__ xyz, float* __restrict__ v0) {
  int idx = blockIdx.x * blockDim.x + threadIdx.x;
  if (idx >= LRES * 9) return;
  int l = idx / 9, x = idx % 3;
  v0[idx] = xyz[idx] - xyz[l * 9 + 3 + x];
}

// ---------------- fused SE3 layer (feat-build + WMMA MLP + masked aggregation) ----------------
// w1T: N-major [64][DIN_PAD]; w2T: N-major [DOUT_PAD][64]
template <int S_IN, int V_IN, int S_OUT, int V_OUT, int DIN_PAD, int DOUT, int DOUT_PAD>
__global__ void se3_kernel(const float* __restrict__ h_in, const float* __restrict__ vin,
                           const half_t* __restrict__ edge, const float* __restrict__ mask,
                           const float* __restrict__ rmat, const float* __restrict__ dhat,
                           const half_t* __restrict__ w1T, const float* __restrict__ b1,
                           const half_t* __restrict__ w2T, const float* __restrict__ b2,
                           const float* __restrict__ Ws, const float* __restrict__ bs,
                           const float* __restrict__ Wv,
                           float* __restrict__ h_out, float* __restrict__ v_out) {
  extern __shared__ char smem[];
  constexpr int AGG = S_OUT + V_OUT * 3;
  half_t* Abase = (half_t*)smem;                                  // 4*16*DIN_PAD
  half_t* Hbase = Abase + 4 * 16 * DIN_PAD;                       // 4*16*64
  float*  Obase = (float*)(Hbase + 4 * 16 * 64);                  // 4*16*DOUT_PAD
  float*  Vbase = Obase + 4 * 16 * DOUT_PAD;                      // 4*16*V_IN*3
  float*  Dbase = Vbase + 4 * 16 * V_IN * 3;                      // 4*16*3
  float*  Mbase = Dbase + 4 * 16 * 3;                             // 4*16
  float*  agg   = Mbase + 4 * 16;                                 // AGG
  half_t* hi_l  = (half_t*)(agg + AGG);                           // S_IN
  int tid = threadIdx.x, lane = tid & 31, wave = tid >> 5;
  int i = blockIdx.x;
  half_t* Aw = Abase + wave * 16 * DIN_PAD;
  half_t* Hw = Hbase + wave * 16 * 64;
  float*  Ow = Obase + wave * 16 * DOUT_PAD;
  float*  Vw = Vbase + wave * 16 * V_IN * 3;
  float*  Dw = Dbase + wave * 16 * 3;
  float*  Mw = Mbase + wave * 16;
  if (tid < AGG) agg[tid] = 0.f;
  if (tid < S_IN) hi_l[tid] = (half_t)h_in[(size_t)i * S_IN + tid];
  __syncthreads();
  for (int jt = wave; jt < LRES / 16; jt += 4) {
    int j0 = jt * 16;
    for (int idx = lane; idx < 16; idx += 32) Mw[idx] = mask[(size_t)i * LRES + j0 + idx];
    for (int idx = lane; idx < 16 * 3; idx += 32)
      Dw[idx] = dhat[((size_t)i * LRES + j0 + idx / 3) * 3 + (idx % 3)];
    for (int idx = lane; idx < 16 * V_IN * 3; idx += 32) {
      int jj = idx / (V_IN * 3), rem = idx % (V_IN * 3);
      Vw[idx] = vin[(size_t)(j0 + jj) * V_IN * 3 + rem];
    }
    __syncthreads();
    for (int idx = lane; idx < 16 * DIN_PAD; idx += 32) {
      int row = idx / DIN_PAD, col = idx % DIN_PAD;
      int j = j0 + row;
      half_t val;
      if (col < S_IN) val = hi_l[col];
      else if (col < 2 * S_IN) val = (half_t)h_in[(size_t)j * S_IN + (col - S_IN)];
      else if (col < 2 * S_IN + DE) val = edge[((size_t)i * LRES + j) * DE + (col - 2 * S_IN)];
      else if (col == 2 * S_IN + DE) val = (half_t)rmat[(size_t)i * LRES + j];
      else if (col < 2 * S_IN + DE + 1 + V_IN) {
        int c = col - (2 * S_IN + DE + 1);
        float s = 0.f;
        for (int x = 0; x < 3; ++x) s += Vw[row * V_IN * 3 + c * 3 + x] * Dw[row * 3 + x];
        val = (half_t)s;
      } else val = (half_t)0.f;
      Aw[idx] = val;
    }
    __syncthreads();
    // hid = elu(feat @ W1 + b1)
    v8f acc1[4] = {};
    for (int kt = 0; kt < DIN_PAD / 32; ++kt) {
      v16h a = frag_a(Aw, DIN_PAD, kt * 32, lane);
#pragma unroll
      for (int nt = 0; nt < 4; ++nt) {
        v16h bfr = frag_bT(w1T, DIN_PAD, kt * 32, nt * 16, lane);
        acc1[nt] = wmma16(a, bfr, acc1[nt]);
      }
    }
    {
      int moff = (lane < 16) ? 0 : 8, nloc = lane & 15;
#pragma unroll
      for (int nt = 0; nt < 4; ++nt) {
        int col = nt * 16 + nloc;
        float bv = b1[col];
#pragma unroll
        for (int rr = 0; rr < 8; ++rr)
          Hw[(moff + rr) * 64 + col] = (half_t)eluf(acc1[nt][rr] + bv);
      }
    }
    __syncthreads();
    // out = hid @ W2 + b2
    v8f acc2[DOUT_PAD / 16] = {};
    for (int kt = 0; kt < 2; ++kt) {
      v16h a = frag_a(Hw, 64, kt * 32, lane);
#pragma unroll
      for (int nt = 0; nt < DOUT_PAD / 16; ++nt) {
        v16h bfr = frag_bT(w2T, 64, kt * 32, nt * 16, lane);
        acc2[nt] = wmma16(a, bfr, acc2[nt]);
      }
    }
    {
      int moff = (lane < 16) ? 0 : 8, nloc = lane & 15;
#pragma unroll
      for (int nt = 0; nt < DOUT_PAD / 16; ++nt) {
        int col = nt * 16 + nloc;
        float bv = (col < DOUT) ? b2[col] : 0.f;
#pragma unroll
        for (int rr = 0; rr < 8; ++rr)
          Ow[(moff + rr) * DOUT_PAD + col] = acc2[nt][rr] + bv;
      }
    }
    __syncthreads();
    // masked aggregation over this 16-j tile
    for (int idx = lane; idx < AGG; idx += 32) {
      float s = 0.f;
      if (idx < S_OUT) {
        for (int jj = 0; jj < 16; ++jj) s += Mw[jj] * Ow[jj * DOUT_PAD + idx];
      } else {
        int q = idx - S_OUT, vo = q / 3, x = q % 3;
        for (int jj = 0; jj < 16; ++jj) {
          float mval = Mw[jj];
          if (mval != 0.f) {
            float t = Ow[jj * DOUT_PAD + S_OUT + vo] * Dw[jj * 3 + x];
            for (int c = 0; c < V_IN; ++c)
              t += Ow[jj * DOUT_PAD + S_OUT + V_OUT + vo * V_IN + c] * Vw[jj * V_IN * 3 + c * 3 + x];
            s += mval * t;
          }
        }
      }
      atomicAdd(&agg[idx], s);
    }
    __syncthreads();
  }
  __syncthreads();
  if (tid < S_OUT) {
    float acc = bs[tid] + agg[tid];
    for (int c = 0; c < S_IN; ++c) acc += h_in[(size_t)i * S_IN + c] * Ws[c * S_OUT + tid];
    h_out[(size_t)i * S_OUT + tid] = eluf(acc);
  }
  if (tid < V_OUT * 3) {
    int vo = tid / 3, x = tid % 3;
    float acc = agg[S_OUT + tid];
    for (int c = 0; c < V_IN; ++c) acc += vin[(size_t)i * V_IN * 3 + c * 3 + x] * Wv[c * V_OUT + vo];
    v_out[(size_t)i * V_OUT * 3 + vo * 3 + x] = acc;
  }
}

// ---------------- finalize: state + xyz_new ----------------
__global__ void final_kernel(const float* __restrict__ h2, const float* __restrict__ v2,
                             const float* __restrict__ xyz, float* __restrict__ out) {
  int i = blockIdx.x, t = threadIdx.x; // 32 threads
  out[(size_t)i * DSTATE + t] = h2[(size_t)i * DSTATE + t];
  if (t < 9) {
    int atom = t / 3, x = t % 3;
    float ca = xyz[i * 9 + 3 + x] + v2[i * 9 + 3 + x];
    float val = (atom == 0) ? ca + v2[i * 9 + x] : (atom == 1) ? ca : ca + v2[i * 9 + 6 + x];
    out[(size_t)LRES * DSTATE + i * 9 + atom * 3 + x] = val;
  }
}

extern "C" void kernel_launch(void* const* d_in, const int* in_sizes, int n_in,
                              void* d_out, int out_size, void* d_ws, size_t ws_size,
                              hipStream_t stream) {
  // inputs in setup_inputs() insertion order (nested dicts flattened depth-first)
  const float* xyz   = (const float*)d_in[0];
  const float* msa   = (const float*)d_in[1];
  const float* pair  = (const float*)d_in[2];
  const float* seq   = (const float*)d_in[3];
  const int*   aa    = (const int*)d_in[4];
  const float* lnms  = (const float*)d_in[5];
  const float* lnmb  = (const float*)d_in[6];
  const float* lnps  = (const float*)d_in[7];
  const float* lnpb  = (const float*)d_in[8];
  const float* q_w   = (const float*)d_in[9];
  const float* q_b   = (const float*)d_in[10];
  const float* k_w   = (const float*)d_in[11];
  /* k_b d_in[12] cancels under softmax */
  const float* node_w = (const float*)d_in[13];
  const float* node_b = (const float*)d_in[14];
  const float* nlns  = (const float*)d_in[15];
  const float* nlnb  = (const float*)d_in[16];
  const float* edge_w = (const float*)d_in[17];
  const float* edge_b = (const float*)d_in[18];
  const float* elns  = (const float*)d_in[19];
  const float* elnb  = (const float*)d_in[20];
  const float* W1_0 = (const float*)d_in[21]; const float* b1_0 = (const float*)d_in[22];
  const float* W2_0 = (const float*)d_in[23]; const float* b2_0 = (const float*)d_in[24];
  const float* Ws_0 = (const float*)d_in[25]; const float* bs_0 = (const float*)d_in[26];
  const float* Wv_0 = (const float*)d_in[27];
  const float* W1_1 = (const float*)d_in[28]; const float* b1_1 = (const float*)d_in[29];
  const float* W2_1 = (const float*)d_in[30]; const float* b2_1 = (const float*)d_in[31];
  const float* Ws_1 = (const float*)d_in[32]; const float* bs_1 = (const float*)d_in[33];
  const float* Wv_1 = (const float*)d_in[34];
  float* out = (float*)d_out;

  char* ws = (char*)d_ws;
  size_t off = 0;
  auto alloc = [&](size_t bytes) -> void* {
    void* p = ws + off;
    off = (off + bytes + 255) & ~(size_t)255;
    return p;
  };
  half_t* msa_ln = (half_t*)alloc((size_t)NSEQ * LRES * DM * 2);
  half_t* qh     = (half_t*)alloc((size_t)LRES * DM * 2);
  float*  tbuf   = (float*)alloc((size_t)LRES * DM * 4);
  float*  pooled = (float*)alloc((size_t)LRES * DM * 4);
  float*  h0     = (float*)alloc((size_t)LRES * DN * 4);
  half_t* edge   = (half_t*)alloc((size_t)LRES * LRES * DE * 2);
  float*  rmat   = (float*)alloc((size_t)LRES * LRES * 4);
  float*  dhat   = (float*)alloc((size_t)LRES * LRES * 3 * 4);
  float*  maskb  = (float*)alloc((size_t)LRES * LRES * 4);
  float*  v0     = (float*)alloc((size_t)LRES * 9 * 4);
  float*  h1     = (float*)alloc((size_t)LRES * 16 * 4);
  float*  v1b    = (float*)alloc((size_t)LRES * 16 * 3 * 4);
  float*  h2     = (float*)alloc((size_t)LRES * 32 * 4);
  float*  v2b    = (float*)alloc((size_t)LRES * 9 * 4);
  half_t* qw16   = (half_t*)alloc((size_t)DM * DM * 2);   // row-major [K=d][N=e]
  half_t* kw16   = (half_t*)alloc((size_t)DM * DM * 2);   // row-major [K=e][N=d] (= k_w^T layout)
  half_t* ewT    = (half_t*)alloc((size_t)DE * DP * 2);   // N-major [64][288]
  half_t* w1_0T  = (half_t*)alloc((size_t)64 * 192 * 2);  // N-major [64][192]
  half_t* w2_0T  = (half_t*)alloc((size_t)80 * 64 * 2);   // N-major [80][64]
  half_t* w1_1T  = (half_t*)alloc((size_t)64 * 128 * 2);  // N-major [64][128]
  half_t* w2_1T  = (half_t*)alloc((size_t)96 * 64 * 2);   // N-major [96][64]
  (void)ws_size; (void)in_sizes; (void)n_in; (void)out_size;

  auto cdiv = [](int a, int b) { return (a + b - 1) / b; };

  // 1) msa LayerNorm -> f16
  ln_rows_f16_kernel<<<NSEQ * LRES, 128, 0, stream>>>(msa, lnms, lnmb, msa_ln, DM);
  // 2) weight conversions
  cvt_f16_kernel<<<cdiv(DM * DM, 128), 128, 0, stream>>>(q_w, qw16, DM * DM);
  // B for t-GEMM is k_w^T row-major [e][d]  == transpose of k_w
  cvt_f16_T_kernel<<<cdiv(DM * DM, 128), 128, 0, stream>>>(k_w, kw16, DM, DM);
  cvt_f16_padT_kernel<<<cdiv(DE * DP, 128), 128, 0, stream>>>(edge_w, ewT, DP, DE, DP, DE);
  cvt_f16_padT_kernel<<<cdiv(64 * 192, 128), 128, 0, stream>>>(W1_0, w1_0T, 192, 64, 192, 64);
  cvt_f16_padT_kernel<<<cdiv(80 * 64, 128), 128, 0, stream>>>(W2_0, w2_0T, 64, 80, 64, 80);
  cvt_f16_padT_kernel<<<cdiv(64 * 128, 128), 128, 0, stream>>>(W1_1, w1_1T, 113, 64, 128, 64);
  cvt_f16_padT_kernel<<<cdiv(96 * 64, 128), 128, 0, stream>>>(W2_1, w2_1T, 64, 83, 64, 96);
  // 3) q = (msa_ln[0] @ q_w + q_b) * dm^-0.5   (WMMA)
  {
    dim3 g(DM / 64, LRES / 64);
    gemm_f16_kernel<<<g, 128, 0, stream>>>(msa_ln, qw16, q_b, rsqrtf((float)DM),
                                           LRES, DM, DM, nullptr, qh);
  }
  // 4) t = q @ k_w^T  (WMMA); k_b cancels under softmax
  {
    dim3 g(DM / 64, LRES / 64);
    gemm_f16_kernel<<<g, 128, 0, stream>>>(qh, kw16, nullptr, 1.f,
                                           LRES, DM, DM, tbuf, nullptr);
  }
  // 5) attention pooling over msa rows
  att_pool_kernel<<<LRES, 64, 0, stream>>>(msa_ln, tbuf, pooled);
  // 6) node MLP + LN
  node_kernel<<<LRES, 64, 0, stream>>>(pooled, seq, node_w, node_b, nlns, nlnb, h0);
  // 7) fused pair-LN + edge GEMM + elu + LN  (WMMA)
  {
    size_t shm = (size_t)64 * DP * 2 + (size_t)64 * 64 * 4 + 128 * 4;
    edge_kernel<<<(LRES * LRES) / 64, 128, shm, stream>>>(pair, lnps, lnpb, ewT,
                                                          edge_b, elns, elnb, edge);
  }
  // 8) geometry + kNN mask
  geom_kernel<<<LRES, 128, 0, stream>>>(xyz, aa, rmat, dhat, maskb);
  vinit_kernel<<<cdiv(LRES * 9, 128), 128, 0, stream>>>(xyz, v0);
  // 9) SE3 layer 1: s 64->16, v 3->16, din=192, dout=80
  {
    size_t shm = (size_t)4 * 16 * 192 * 2 + 4 * 16 * 64 * 2 + (size_t)4 * 16 * 80 * 4 +
                 (size_t)4 * 16 * 3 * 3 * 4 + 4 * 16 * 3 * 4 + 4 * 16 * 4 +
                 (16 + 48) * 4 + 64 * 2 + 16;
    se3_kernel<64, 3, 16, 16, 192, 80, 80><<<LRES, 128, shm, stream>>>(
        h0, v0, edge, maskb, rmat, dhat, w1_0T, b1_0, w2_0T, b2_0, Ws_0, bs_0, Wv_0, h1, v1b);
  }
  // 10) SE3 layer 2: s 16->32, v 16->3, din=113(pad128), dout=83(pad96)
  {
    size_t shm = (size_t)4 * 16 * 128 * 2 + 4 * 16 * 64 * 2 + (size_t)4 * 16 * 96 * 4 +
                 (size_t)4 * 16 * 16 * 3 * 4 + 4 * 16 * 3 * 4 + 4 * 16 * 4 +
                 (32 + 9) * 4 + 16 * 2 + 16;
    se3_kernel<16, 16, 32, 3, 128, 83, 96><<<LRES, 128, shm, stream>>>(
        h1, v1b, edge, maskb, rmat, dhat, w1_1T, b1_1, w2_1T, b2_1, Ws_1, bs_1, Wv_1, h2, v2b);
  }
  // 11) outputs: state (L,32) then xyz_new (L,3,3)
  final_kernel<<<LRES, 32, 0, stream>>>(h2, v2b, xyz, out);
}
